// MultiHeadCausalAttention_57664230916135
// MI455X (gfx1250) — matrix operand order repaired and checked by reading, not compiled, not run on hardware
//
#include <hip/hip_runtime.h>

// ---------------------------------------------------------------------------
// MI455X (gfx1250) multi-head causal attention, f16 WMMA pipeline.
// B=4, S=2048, D=1024, H=16, DK=64.  ~137 GFLOP, compute-bound.
// GEMMs: ping-pong register double-buffered WMMA loops (no rotation movs).
// Attention: flash w/ online softmax, double-buffered K operands.
// ---------------------------------------------------------------------------

typedef __attribute__((ext_vector_type(16))) _Float16 v16h;
typedef __attribute__((ext_vector_type(8)))  _Float16 v8h;
typedef __attribute__((ext_vector_type(8)))  float    v8f;
typedef __attribute__((ext_vector_type(4)))  float    v4f;
typedef __attribute__((ext_vector_type(4)))  _Float16 v4h;

#define DEV static __device__ __forceinline__

static constexpr int Bz = 4;
static constexpr int Sq = 2048;
static constexpr int Dm = 1024;
static constexpr int Hh = 16;
static constexpr int DK = 64;
static constexpr int Mrows = Bz * Sq;       // 8192 tokens

// ---------------------------------------------------------------------------
// WMMA operand loaders (wave32). ISA 7.12.2 layouts:
//  A 16x32 f16: lanes 0-15 row m=lane, elems 0..7 = K0..7, 8..15 = K16..23;
//               lanes 16-31 row m=lane-16, elems = K8..15 / K24..31.
//  B 32x16 f16: lanes 0-15 col n=lane, elems 0..7 = K0..7, 8..15 = K16..23;
//               lanes 16-31 col n=lane-16, elems = K8..15 / K24..31.
// ---------------------------------------------------------------------------
DEV v16h load_a16x32(const _Float16* base, int ld) {
  const int lane = threadIdx.x & 31;
  const int r    = lane & 15;
  const int kb   = (lane >> 4) << 3;                 // 0 or 8
  const _Float16* p = base + r * ld + kb;
  v8h lo = *(const v8h*)(p);
  v8h hi = *(const v8h*)(p + 16);
  v16h out;
#pragma unroll
  for (int i = 0; i < 8; ++i) { out[i] = lo[i]; out[i + 8] = hi[i]; }
  return out;
}

DEV v16h load_b32x16(const _Float16* base, int ld) {
  const int lane = threadIdx.x & 31;
  const int n    = lane & 15;
  const int kb   = (lane >> 4) << 3;                 // 0 or 8
  const _Float16* p = base + n * ld + kb;
  v8h lo = *(const v8h*)(p);
  v8h hi = *(const v8h*)(p + 16);
  v16h out;
#pragma unroll
  for (int i = 0; i < 8; ++i) { out[i] = lo[i]; out[i + 8] = hi[i]; }
  return out;
}

// Per-lane prefetch of a 16-row tile region (emits global_prefetch_b8).
DEV void prefetch_tile(const _Float16* base, int ld) {
  const int lane = threadIdx.x & 31;
  __builtin_prefetch(base + (lane & 15) * ld + ((lane >> 4) << 4), 0, 3);
}

DEV v8f wmma_f16(v16h a, v16h b, v8f c) {
  return __builtin_amdgcn_wmma_f32_16x16x32_f16(false, a, false, b,
                                                (short)0, c, false, false);
}

// ---------------------------------------------------------------------------
// fp32 -> f16 convert (vectorized x4)
// ---------------------------------------------------------------------------
__global__ void __launch_bounds__(256)
f32_to_f16_kernel(const float* __restrict__ src, _Float16* __restrict__ dst, int n4) {
  int i = blockIdx.x * 256 + threadIdx.x;
  if (i < n4) {
    v4f v = *(const v4f*)(src + (long)i * 4);
    v4h h;
#pragma unroll
    for (int j = 0; j < 4; ++j) h[j] = (_Float16)v[j];
    *(v4h*)(dst + (long)i * 4) = h;
  }
}

// ---------------------------------------------------------------------------
// Ping-pong K-loop building blocks: set = {a0, a1, b[4]} for a 32-wide K step
// over a 32-row x 64-col output tile (8 WMMAs per consume).
// ---------------------------------------------------------------------------
#define LOAD_SET(a0v, a1v, bv, kk)                                       \
  do {                                                                   \
    a0v = load_a16x32(A0 + (kk), Dm);                                    \
    a1v = load_a16x32(A1 + (kk), Dm);                                    \
    _Pragma("unroll")                                                    \
    for (int nt_ = 0; nt_ < 4; ++nt_)                                    \
      bv[nt_] = load_b32x16(Bp + nt_ * 16 * Dm + (kk), Dm);              \
  } while (0)

#define WMMA_SET(a0v, a1v, bv)                                           \
  do {                                                                   \
    _Pragma("unroll")                                                    \
    for (int nt_ = 0; nt_ < 4; ++nt_) {                                  \
      acc[0][nt_] = wmma_f16(a0v, bv[nt_], acc[0][nt_]);                 \
      acc[1][nt_] = wmma_f16(a1v, bv[nt_], acc[1][nt_]);                 \
    }                                                                    \
  } while (0)

// ---------------------------------------------------------------------------
// Projection GEMM: Y = Xh @ Wh^T.  One wave: 32 tokens x one head (64 outs).
// transposed==0 -> [B,H,S,DK];  ==1 -> [B,H,DK,S] (V, so PV B-operand loads
// in the attention kernel are contiguous over keys).
// ---------------------------------------------------------------------------
__global__ void __launch_bounds__(32)
gemm_proj_f16(const _Float16* __restrict__ Xh, const _Float16* __restrict__ Wh,
              _Float16* __restrict__ dst, int transposed) {
  const int mt   = blockIdx.x;           // 0..255 (32-row tiles)
  const int h    = blockIdx.y;           // 0..15 head
  const int lane = threadIdx.x & 31;
  const int grp  = lane >> 4;
  const int n15  = lane & 15;
  const int row0 = mt * 32;

  const _Float16* A0 = Xh + (long)row0 * Dm;
  const _Float16* A1 = Xh + (long)(row0 + 16) * Dm;
  const _Float16* Bp = Wh + (long)(h * DK) * Dm;

  v8f acc[2][4] = {};
  v16h a0A, a1A, bA[4];
  v16h a0B, a1B, bB[4];
  LOAD_SET(a0A, a1A, bA, 0);
  LOAD_SET(a0B, a1B, bB, 32);

  int k = 0;
  for (; k + 64 < Dm; k += 64) {
    prefetch_tile(A0 + k + 160, Dm);
    prefetch_tile(A1 + k + 160, Dm);
    WMMA_SET(a0A, a1A, bA);
    LOAD_SET(a0A, a1A, bA, k + 64);
    WMMA_SET(a0B, a1B, bB);
    LOAD_SET(a0B, a1B, bB, k + 96);
  }
  WMMA_SET(a0A, a1A, bA);
  WMMA_SET(a0B, a1B, bB);

#pragma unroll
  for (int ms = 0; ms < 2; ++ms) {
#pragma unroll
    for (int nt = 0; nt < 4; ++nt) {
#pragma unroll
      for (int r = 0; r < 8; ++r) {
        const int m   = r + grp * 8;
        const int row = row0 + ms * 16 + m;    // global token
        const int b   = row >> 11;             // / 2048
        const int s   = row & (Sq - 1);
        const int dk  = nt * 16 + n15;
        const long bh = (long)b * Hh + h;
        const long idx = transposed ? (bh * DK + dk) * Sq + s
                                    : (bh * Sq + s) * DK + dk;
        dst[idx] = (_Float16)acc[ms][nt][r];
      }
    }
  }
}

// ---------------------------------------------------------------------------
// Flash attention, one wave per (b, h, 16-query tile). Online softmax with
// per-lane row-replicated stats; K operands double-buffered (next chunk's K
// loads complete during this chunk's softmax); V operands issued early.
// ---------------------------------------------------------------------------
__global__ void __launch_bounds__(32)
flash_attn_kernel(const _Float16* __restrict__ Qh,   // [B*H, S, DK]
                  const _Float16* __restrict__ Kh,   // [B*H, S, DK]
                  const _Float16* __restrict__ Vt,   // [B*H, DK, S]
                  _Float16* __restrict__ Oh) {       // [B, S, H*DK]
  __shared__ _Float16 pLds[16 * 32];

  const int qt   = blockIdx.x;            // 0..127
  const int h    = blockIdx.y;            // 0..15
  const int b    = blockIdx.z;            // 0..3
  const long bh  = (long)b * Hh + h;
  const int lane = threadIdx.x & 31;
  const int grp  = lane >> 4;
  const int n15  = lane & 15;
  const int q0   = qt * 16;
  const float scale = 0.125f;             // 1/sqrt(64)

  const _Float16* Qbase = Qh + (bh * Sq + q0) * DK;
  const v16h qa0 = load_a16x32(Qbase, DK);        // dk 0..31
  const v16h qa1 = load_a16x32(Qbase + 32, DK);   // dk 32..63

  v8f o[4] = {};
  float rowmax[8], rowsum[8];
#pragma unroll
  for (int r = 0; r < 8; ++r) { rowmax[r] = -__builtin_inff(); rowsum[r] = 0.f; }

  // K B-operands for one 32-key chunk: [t*2 + khalf], t = key sub-tile.
  v16h kb[4];
  {
    const _Float16* Kb = Kh + bh * Sq * DK;
    kb[0] = load_b32x16(Kb, DK);
    kb[1] = load_b32x16(Kb + 32, DK);
    kb[2] = load_b32x16(Kb + 16 * DK, DK);
    kb[3] = load_b32x16(Kb + 16 * DK + 32, DK);
  }

  const int nchunks = (q0 + 16 + 31) >> 5;        // keys up to q0+15 inclusive
  for (int c = 0; c < nchunks; ++c) {
    const int kbase = c * 32;

    // ---- scores: S = Q K^T, two 16x16 key sub-tiles, K-dim = DK = 64 ----
    v8f s[2];
#pragma unroll
    for (int t = 0; t < 2; ++t) {
      v8f cacc = {};
      cacc = wmma_f16(qa0, kb[t * 2 + 0], cacc);
      cacc = wmma_f16(qa1, kb[t * 2 + 1], cacc);
      s[t] = cacc;
    }

    // ---- issue V loads now; they complete during the softmax below ----
    v16h vb[4];
#pragma unroll
    for (int nt = 0; nt < 4; ++nt)
      vb[nt] = load_b32x16(Vt + (bh * DK + nt * 16) * Sq + kbase, Sq);

    // ---- issue next chunk's K loads (consumed next iteration) ----
    v16h kbn[4];
    const bool more = (c + 1 < nchunks);
    if (more) {
      const int kn = kbase + 32;
      const _Float16* Kb = Kh + (bh * Sq + kn) * DK;
      kbn[0] = load_b32x16(Kb, DK);
      kbn[1] = load_b32x16(Kb + 32, DK);
      kbn[2] = load_b32x16(Kb + 16 * DK, DK);
      kbn[3] = load_b32x16(Kb + 16 * DK + 32, DK);
      // prefetch chunk c+2 lines
      prefetch_tile(Kh + (bh * Sq + kn + 32) * DK, DK);
      prefetch_tile(Kh + (bh * Sq + kn + 48) * DK, DK);
#pragma unroll
      for (int nt = 0; nt < 4; ++nt)
        prefetch_tile(Vt + (bh * DK + nt * 16) * Sq + kn, Sq);
    }

    // ---- causal mask + scale + online softmax update ----
#pragma unroll
    for (int r = 0; r < 8; ++r) {
      const int m    = r + grp * 8;
      const int qrow = q0 + m;
#pragma unroll
      for (int t = 0; t < 2; ++t) {
        const int kcol = kbase + t * 16 + n15;
        const float v  = s[t][r] * scale;
        s[t][r] = (kcol > qrow) ? -__builtin_inff() : v;
      }
      float cm = fmaxf(s[0][r], s[1][r]);
#pragma unroll
      for (int off = 8; off >= 1; off >>= 1)
        cm = fmaxf(cm, __shfl_xor(cm, off, 16));
      const float nm    = fmaxf(rowmax[r], cm);
      const float alpha = (rowmax[r] == -__builtin_inff())
                              ? 0.f : __expf(rowmax[r] - nm);
      float psum = 0.f;
#pragma unroll
      for (int t = 0; t < 2; ++t) {
        const float p = __expf(s[t][r] - nm);      // exp(-inf) = 0 for masked
        s[t][r] = p;
        psum += p;
      }
#pragma unroll
      for (int off = 8; off >= 1; off >>= 1)
        psum += __shfl_xor(psum, off, 16);
      rowsum[r] = rowsum[r] * alpha + psum;
      rowmax[r] = nm;
#pragma unroll
      for (int nt = 0; nt < 4; ++nt) o[nt][r] *= alpha;
    }

    // ---- P: D-layout -> A-layout via LDS (16x32 f16 row-major) ----
    __syncthreads();
#pragma unroll
    for (int r = 0; r < 8; ++r) {
      const int m = r + grp * 8;
      pLds[m * 32 + n15]      = (_Float16)s[0][r];
      pLds[m * 32 + 16 + n15] = (_Float16)s[1][r];
    }
    __syncthreads();
    const v16h pa = load_a16x32(pLds, 32);

    // ---- O += P V (V operands already in registers) ----
#pragma unroll
    for (int nt = 0; nt < 4; ++nt)
      o[nt] = wmma_f16(pa, vb[nt], o[nt]);
    __syncthreads();

    // rotate double-buffered K (copies land after PV issue; loads long done)
    if (more) {
#pragma unroll
      for (int i = 0; i < 4; ++i) kb[i] = kbn[i];
    }
  }

  // ---- normalize and store to [B, S, H*DK] f16 ----
#pragma unroll
  for (int r = 0; r < 8; ++r) {
    const int m     = r + grp * 8;
    const float inv = 1.f / rowsum[r];
    const int srow  = q0 + m;
#pragma unroll
    for (int nt = 0; nt < 4; ++nt) {
      const long idx = ((long)b * Sq + srow) * Dm + h * DK + nt * 16 + n15;
      Oh[idx] = (_Float16)(o[nt][r] * inv);
    }
  }
}

// ---------------------------------------------------------------------------
// Output GEMM: out = Ah @ Wo^T + b_o, fp32 store; ping-pong pipelined.
// ---------------------------------------------------------------------------
__global__ void __launch_bounds__(32)
gemm_out_f32(const _Float16* __restrict__ Ah, const _Float16* __restrict__ Wh,
             const float* __restrict__ bias, float* __restrict__ out) {
  const int mt   = blockIdx.x;           // 0..255
  const int ng   = blockIdx.y;           // 0..15 (group of 64 cols)
  const int lane = threadIdx.x & 31;
  const int grp  = lane >> 4;
  const int n15  = lane & 15;
  const int row0 = mt * 32;

  const _Float16* A0 = Ah + (long)row0 * Dm;
  const _Float16* A1 = Ah + (long)(row0 + 16) * Dm;
  const _Float16* Bp = Wh + (long)(ng * 64) * Dm;

  v8f acc[2][4] = {};
  v16h a0A, a1A, bA[4];
  v16h a0B, a1B, bB[4];
  LOAD_SET(a0A, a1A, bA, 0);
  LOAD_SET(a0B, a1B, bB, 32);

  int k = 0;
  for (; k + 64 < Dm; k += 64) {
    prefetch_tile(A0 + k + 160, Dm);
    prefetch_tile(A1 + k + 160, Dm);
    WMMA_SET(a0A, a1A, bA);
    LOAD_SET(a0A, a1A, bA, k + 64);
    WMMA_SET(a0B, a1B, bB);
    LOAD_SET(a0B, a1B, bB, k + 96);
  }
  WMMA_SET(a0A, a1A, bA);
  WMMA_SET(a0B, a1B, bB);

#pragma unroll
  for (int ms = 0; ms < 2; ++ms) {
#pragma unroll
    for (int nt = 0; nt < 4; ++nt) {
#pragma unroll
      for (int r = 0; r < 8; ++r) {
        const int m   = r + grp * 8;
        const int col = ng * 64 + nt * 16 + n15;
        out[(long)(row0 + ms * 16 + m) * Dm + col] = acc[ms][nt][r] + bias[col];
      }
    }
  }
}

// ---------------------------------------------------------------------------
// Host launcher
// ---------------------------------------------------------------------------
extern "C" void kernel_launch(void* const* d_in, const int* in_sizes, int n_in,
                              void* d_out, int out_size, void* d_ws, size_t ws_size,
                              hipStream_t stream) {
  (void)in_sizes; (void)n_in; (void)out_size; (void)ws_size;

  const float* x  = (const float*)d_in[0];
  const float* wq = (const float*)d_in[1];
  const float* wk = (const float*)d_in[2];
  const float* wv = (const float*)d_in[3];
  const float* wo = (const float*)d_in[4];
  const float* bo = (const float*)d_in[5];
  float* out = (float*)d_out;

  const size_t NX = (size_t)Mrows * Dm;   // 8,388,608
  const size_t NW = (size_t)Dm * Dm;      // 1,048,576

  char* ws = (char*)d_ws;
  _Float16* Xh  = (_Float16*)ws;                 ws += NX * sizeof(_Float16);
  _Float16* Wqh = (_Float16*)ws;                 ws += NW * sizeof(_Float16);
  _Float16* Wkh = (_Float16*)ws;                 ws += NW * sizeof(_Float16);
  _Float16* Wvh = (_Float16*)ws;                 ws += NW * sizeof(_Float16);
  _Float16* Woh = (_Float16*)ws;                 ws += NW * sizeof(_Float16);
  _Float16* Qh  = (_Float16*)ws;                 ws += NX * sizeof(_Float16);
  _Float16* Kh  = (_Float16*)ws;                 ws += NX * sizeof(_Float16);
  _Float16* Vt  = (_Float16*)ws;                 ws += NX * sizeof(_Float16);
  _Float16* Ah  = (_Float16*)ws;                 ws += NX * sizeof(_Float16);

  f32_to_f16_kernel<<<(int)((NX / 4 + 255) / 256), 256, 0, stream>>>(x,  Xh,  (int)(NX / 4));
  f32_to_f16_kernel<<<(int)((NW / 4 + 255) / 256), 256, 0, stream>>>(wq, Wqh, (int)(NW / 4));
  f32_to_f16_kernel<<<(int)((NW / 4 + 255) / 256), 256, 0, stream>>>(wk, Wkh, (int)(NW / 4));
  f32_to_f16_kernel<<<(int)((NW / 4 + 255) / 256), 256, 0, stream>>>(wv, Wvh, (int)(NW / 4));
  f32_to_f16_kernel<<<(int)((NW / 4 + 255) / 256), 256, 0, stream>>>(wo, Woh, (int)(NW / 4));

  dim3 gproj(Mrows / 32, Hh);             // 256 x 16 waves
  gemm_proj_f16<<<gproj, 32, 0, stream>>>(Xh, Wqh, Qh, 0);
  gemm_proj_f16<<<gproj, 32, 0, stream>>>(Xh, Wkh, Kh, 0);
  gemm_proj_f16<<<gproj, 32, 0, stream>>>(Xh, Wvh, Vt, 1);   // V transposed

  dim3 gattn(Sq / 16, Hh, Bz);            // 128 x 16 x 4 waves
  flash_attn_kernel<<<gattn, 32, 0, stream>>>(Qh, Kh, Vt, Ah);

  dim3 gout(Mrows / 32, Hh);
  gemm_out_f32<<<gout, 32, 0, stream>>>(Ah, Woh, bo, out);
}
